// GCNEmbedding2_4724464025863
// MI455X (gfx1250) — matrix operand order repaired and checked by reading, not compiled, not run on hardware
//
#include <hip/hip_runtime.h>

// ---------------------------------------------------------------------------
// GCN-embedding forward for MI455X (gfx1250, wave32, WMMA).
// Convs decomposed per filter tap: out = sum_k W_k(64xCin) * X(:, l+k-K/2),
// run on v_wmma_f32_16x16x32_f16 with fp32 accumulators. The LDS input tile
// is stored transposed ([l][ci]) so B fragments are contiguous ds_load_b128s.
// BatchNorm stats come from the fp32 accumulators and are folded into
// scale/shift applied when the next conv loads its input.
// ---------------------------------------------------------------------------

typedef _Float16 h16;
typedef __attribute__((ext_vector_type(16))) _Float16 v16h;
typedef __attribute__((ext_vector_type(8)))  _Float16 v8h;
typedef __attribute__((ext_vector_type(8)))  float    v8f;

#define BB   32
#define NN   64
#define CINV 8
#define LL   256
#define MT   2048            // BB*NN samples
#define XOFF 4               // left halo rows (max K/2 = 3 for K=7)
#define XROWS 264            // 256 + 8 halo rows
#define XSTR 72              // halves per LDS row: 144B, 16B aligned, offbeat banks
#define EPSV 1e-5f

// ---------------------------------------------------------------------------
// Conv kernel: one workgroup (8 wave32) per sample m.
//   out[c][l] = bias[c] + sum_{k,ci} A[c][k*CPAD+ci] * xin'[ci][l+k-K/2]
// xin' = optional (scale*x+shift) then optional relu. Also accumulates
// per-(graph,channel) sum/sumsq of the fp32 raw output for BatchNorm.
// K: taps; CPSH: log2 of padded channel count; NK: K*CPAD/32 WMMA steps.
// ---------------------------------------------------------------------------
template<int K, int CPSH, int NK>
__global__ __launch_bounds__(256)
void conv_wmma(const h16* __restrict__ xin, int cin,
               const float* __restrict__ scaleIn, const float* __restrict__ shiftIn,
               int applyNorm, int applyRelu,
               const h16* __restrict__ Aw,
               const float* __restrict__ bias,
               h16* __restrict__ outp,
               float* __restrict__ statSum, float* __restrict__ statSq)
{
    constexpr int KDIM = NK * 32;       // padded GEMM K-dimension
    constexpr int CPAD = 1 << CPSH;     // padded input channels per tap

    __shared__ __align__(16) h16 Xs[XROWS * XSTR];   // [l+XOFF][ci], 38 KB

    const int m   = blockIdx.x;         // sample = b*64 + n
    const int b   = m >> 6;
    const int tid = threadIdx.x;

    // Zero the tile with wide stores (halo rows + padded channels stay zero).
    {
        const v8h zv = {};
        v8h* Z = (v8h*)Xs;
        for (int i = tid; i < (XROWS * XSTR) / 8; i += 256) Z[i] = zv;
    }
    __syncthreads();

    // Load input (fp16, 16B vectors), apply folded BN scale/shift (+relu),
    // store transposed into LDS.
    {
        const h16* src = xin + (size_t)m * cin * LL;
        const int nvec = (cin * LL) / 8;
        for (int iv = tid; iv < nvec; iv += 256) {
            const int i = iv << 3;
            const int c = i >> 8, l = i & 255;
            v8h xv = *(const v8h*)(src + i);
            float sc = 1.f, sh = 0.f;
            if (applyNorm) { sc = scaleIn[(b << 6) + c]; sh = shiftIn[(b << 6) + c]; }
#pragma unroll
            for (int j = 0; j < 8; ++j) {
                float v = (float)xv[j] * sc + sh;
                if (applyRelu) v = fmaxf(v, 0.f);
                Xs[(XOFF + l + j) * XSTR + c] = (h16)v;
            }
        }
    }
    __syncthreads();

    const int wave     = tid >> 5;
    const int lane     = tid & 31;
    const int lane16   = lane & 15;
    const int laneHalf = lane >> 4;
    const int n0       = wave << 5;     // each wave owns 32 output columns

    const v8f zero = {0.f, 0.f, 0.f, 0.f, 0.f, 0.f, 0.f, 0.f};
    v8f acc[4][2];
#pragma unroll
    for (int i = 0; i < 4; ++i) { acc[i][0] = zero; acc[i][1] = zero; }

#pragma unroll
    for (int kk = 0; kk < NK; ++kk) {
        const int kbase  = kk << 5;
        const int k      = kbase >> CPSH;          // filter tap
        const int cb     = kbase & (CPAD - 1);     // channel block within tap
        const int rowoff = k - (K >> 1) + XOFF;    // shifted column -> LDS row

        // B fragments: 16 consecutive channels at a fixed column -> 2x b128.
        v16h bf[2];
#pragma unroll
        for (int t = 0; t < 2; ++t) {
            const int l = n0 + (t << 4) + lane16;
            const h16* src = &Xs[(l + rowoff) * XSTR + cb + (laneHalf << 4)];
            v8h b0 = *(const v8h*)(src);
            v8h b1 = *(const v8h*)(src + 8);
            bf[t] = __builtin_shufflevector(b0, b1,
                        0,1,2,3,4,5,6,7,8,9,10,11,12,13,14,15);
        }

        // A fragments: contiguous 16B loads from tap-major fp16 weights.
#pragma unroll
        for (int mt = 0; mt < 4; ++mt) {
            const h16* arow = Aw + (size_t)((mt << 4) + lane16) * KDIM
                                 + kbase + (laneHalf << 4);
            v8h alo = *(const v8h*)(arow);
            v8h ahi = *(const v8h*)(arow + 8);
            v16h af = __builtin_shufflevector(alo, ahi,
                        0,1,2,3,4,5,6,7,8,9,10,11,12,13,14,15);
            acc[mt][0] = __builtin_amdgcn_wmma_f32_16x16x32_f16(
                false, af, false, bf[0], (short)0, acc[mt][0], false, false);
            acc[mt][1] = __builtin_amdgcn_wmma_f32_16x16x32_f16(
                false, af, false, bf[1], (short)0, acc[mt][1], false, false);
        }
    }

    // Epilogue: add bias, store fp16 raw output, accumulate BN stats in fp32.
    const size_t obase = (size_t)m * (64 * LL);
#pragma unroll
    for (int mt = 0; mt < 4; ++mt) {
#pragma unroll
        for (int r = 0; r < 8; ++r) {
            const int c   = (mt << 4) + (laneHalf << 3) + r;
            const float bsv = bias[c];
            float s = 0.f, q = 0.f;
#pragma unroll
            for (int t = 0; t < 2; ++t) {
                const int l = n0 + (t << 4) + lane16;
                float v = acc[mt][t][r] + bsv;
                outp[obase + ((size_t)c << 8) + l] = (h16)v;
                s += v; q += v * v;
            }
#pragma unroll
            for (int off = 8; off >= 1; off >>= 1) {
                s += __shfl_xor(s, off, 32);
                q += __shfl_xor(q, off, 32);
            }
            if (lane16 == 0) {
                atomicAdd(&statSum[(b << 6) + c], s);
                atomicAdd(&statSq [(b << 6) + c], q);
            }
        }
    }
}

// Per-(graph,channel) BN folding: scale = g*rsqrt(var+eps), shift = beta-mean*scale.
__global__ void bn_finalize(const float* __restrict__ sum, const float* __restrict__ sq,
                            const float* __restrict__ gamma, const float* __restrict__ beta,
                            float* __restrict__ scale, float* __restrict__ shift)
{
    int i = blockIdx.x * blockDim.x + threadIdx.x;
    if (i >= BB * 64) return;
    int c = i & 63;
    const float inv = 1.f / (64.f * 256.f);
    float mean = sum[i] * inv;
    float var  = sq[i] * inv - mean * mean;
    float sc   = gamma[c] * rsqrtf(var + EPSV);
    scale[i] = sc;
    shift[i] = beta[c] - mean * sc;
}

// Stats of an already-final fp16 activation buffer (identity-shortcut BN).
__global__ __launch_bounds__(256)
void act_stats(const h16* __restrict__ x, float* __restrict__ sum, float* __restrict__ sq)
{
    __shared__ float rs[256], rq[256];
    const int bc = blockIdx.x;        // b*64 + c
    const int b = bc >> 6, c = bc & 63;
    const int tid = threadIdx.x;
    const h16* base = x + ((size_t)(b << 6) * 64 + c) * LL;   // row (b, n=0, c)
    float s = 0.f, q = 0.f;
    for (int i = tid * 8; i < NN * LL; i += 256 * 8) {
        int n = i >> 8, l = i & 255;
        v8h v = *(const v8h*)(base + (size_t)n * (64 * LL) + l);
#pragma unroll
        for (int j = 0; j < 8; ++j) {
            float f = (float)v[j];
            s += f; q += f * f;
        }
    }
    rs[tid] = s; rq[tid] = q;
    __syncthreads();
    for (int off = 128; off > 0; off >>= 1) {
        if (tid < off) { rs[tid] += rs[tid + off]; rq[tid] += rq[tid + off]; }
        __syncthreads();
    }
    if (tid == 0) { sum[bc] = rs[0]; sq[bc] = rq[0]; }
}

// out = relu(scz*t3+shz + scs*sr+shs), fp16 in/out, 8 halves per thread.
__global__ __launch_bounds__(256)
void residual_add(const h16* __restrict__ t3, const h16* __restrict__ sr,
                  const float* __restrict__ scz, const float* __restrict__ shz,
                  const float* __restrict__ scs, const float* __restrict__ shs,
                  h16* __restrict__ outp)
{
    const size_t i0 = ((size_t)blockIdx.x * 256 + threadIdx.x) * 8;
    if (i0 >= (size_t)MT * 64 * LL) return;
    const int m  = (int)(i0 >> 14);
    const int b  = m >> 6;
    const int c  = (int)((i0 >> 8) & 63);
    const int bc = (b << 6) + c;
    const float az = scz[bc], bz = shz[bc], as = scs[bc], bs2 = shs[bc];
    v8h t = *(const v8h*)(t3 + i0);
    v8h s = *(const v8h*)(sr + i0);
    v8h o;
#pragma unroll
    for (int j = 0; j < 8; ++j) {
        float v = az * (float)t[j] + bz + as * (float)s[j] + bs2;
        o[j] = (h16)fmaxf(v, 0.f);
    }
    *(v8h*)(outp + i0) = o;
}

// pooled[b][n][c] = mean_l x[m][c][l]
__global__ __launch_bounds__(256)
void avg_pool(const h16* __restrict__ x, float* __restrict__ pooled)
{
    const int i = blockIdx.x * 256 + threadIdx.x;   // (m, c)
    if (i >= MT * 64) return;
    const h16* row = x + (size_t)i * LL;
    float s = 0.f;
    for (int l = 0; l < LL; l += 8) {
        v8h v = *(const v8h*)(row + l);
#pragma unroll
        for (int j = 0; j < 8; ++j) s += (float)v[j];
    }
    pooled[i] = s * (1.f / (float)LL);
}

// Tiny GCN head: out[b][f] = sum_{f'} (sum_m (mean_n adj[b][n][m]) * pooled[b][m][f']) * W[f'][f] + gc_b[f]
__global__ __launch_bounds__(64)
void gcn_head(const float* __restrict__ pooled, const float* __restrict__ adj,
              const float* __restrict__ gcw, const float* __restrict__ gcb,
              float* __restrict__ outp)
{
    __shared__ float wm[64], vf[64];
    const int b = blockIdx.x, t = threadIdx.x;
    float s = 0.f;
    for (int n = 0; n < 64; ++n) s += adj[((size_t)((b << 6) + n)) * 64 + t];
    wm[t] = s * (1.f / 64.f);
    __syncthreads();
    float v = 0.f;
    for (int m2 = 0; m2 < 64; ++m2) v += wm[m2] * pooled[(size_t)((b << 6) + m2) * 64 + t];
    vf[t] = v;
    __syncthreads();
    float o = gcb[t];
    for (int f2 = 0; f2 < 64; ++f2) o += vf[f2] * gcw[f2 * 64 + t];
    outp[(b << 6) + t] = o;
}

__global__ void cvt_data(const float* __restrict__ in, h16* __restrict__ out, int n)
{
    int i = blockIdx.x * 256 + threadIdx.x;
    if (i < n) out[i] = (h16)in[i];
}

// fp32 conv weights [64][cin][K] -> fp16 tap-major [64][k*cinPad+ci], zero pad.
__global__ void cvt_weight(const float* __restrict__ w, h16* __restrict__ A,
                           int cin, int cinPad, int K, int kdimPad)
{
    int i = blockIdx.x * 256 + threadIdx.x;
    if (i >= 64 * kdimPad) return;
    int co = i / kdimPad, kd = i - co * kdimPad;
    int k  = kd / cinPad, ci = kd - k * cinPad;
    h16 v = (h16)0.f;
    if (ci < cin) v = (h16)w[(co * cin + ci) * K + k];
    A[i] = v;
}

// ---------------------------------------------------------------------------
extern "C" void kernel_launch(void* const* d_in, const int* in_sizes, int n_in,
                              void* d_out, int out_size, void* d_ws, size_t ws_size,
                              hipStream_t stream)
{
    (void)n_in; (void)out_size; (void)ws_size;

    // ---- input mapping (handles both top-level orderings) ----
    int dataI, adjI, pS;
    if (in_sizes[0] == BB * NN * NN) { adjI = 0; dataI = 1; pS = 3; } // adj,data,idx,params...
    else                              { dataI = 0; adjI = 1; pS = 2; } // data,adj,params...,idx
    const float* data = (const float*)d_in[dataI];
    const float* adj  = (const float*)d_in[adjI];

    struct BP { const float *bes,*bex,*bey,*bez,*bs,*bx,*by,*bz,*gs,*gx,*gy,*gz,*ws,*wx,*wy,*wz; };
    BP bp[4]; int q = pS;
    for (int blk = 0; blk < 4; ++blk) {
        BP p{};
        p.bes = (const float*)d_in[q++];
        p.bex = (const float*)d_in[q++];
        p.bey = (const float*)d_in[q++];
        p.bez = (const float*)d_in[q++];
        if (blk == 0) p.bs = (const float*)d_in[q++];
        p.bx  = (const float*)d_in[q++];
        p.by  = (const float*)d_in[q++];
        p.bz  = (const float*)d_in[q++];
        p.gs  = (const float*)d_in[q++];
        p.gx  = (const float*)d_in[q++];
        p.gy  = (const float*)d_in[q++];
        p.gz  = (const float*)d_in[q++];
        if (blk == 0) p.ws = (const float*)d_in[q++];
        p.wx  = (const float*)d_in[q++];
        p.wy  = (const float*)d_in[q++];
        p.wz  = (const float*)d_in[q++];
        bp[blk] = p;
    }
    const float* gcb = (const float*)d_in[q++];
    const float* gcw = (const float*)d_in[q++];

    // ---- workspace layout ----
    const size_t ACT = (size_t)MT * 64 * LL;          // halves per activation buffer
    h16* P[4];
    P[0] = (h16*)d_ws;
    P[1] = P[0] + ACT; P[2] = P[1] + ACT; P[3] = P[2] + ACT;
    float* stats = (float*)(P[0] + 4 * ACT);          // 16 sets x {sum,sq,scale,shift} x 2048
    auto SUM = [&](int s){ return stats + (size_t)s * 4 * 2048; };
    auto SQ  = [&](int s){ return SUM(s) + 2048; };
    auto SC  = [&](int s){ return SUM(s) + 4096; };
    auto SH  = [&](int s){ return SUM(s) + 6144; };
    float* pooled = stats + 16 * 4 * 2048;            // 131072 floats
    h16* wbase = (h16*)(pooled + MT * 64);

    // converted weights (tap-major): [block][0=wx,1=wy,2=wz,3=ws]
    // kdimPad: blk0 wx 7*32=224, ws 1*32=32; wy 5*64=320; wz 3*64=192; wx 7*64=448
    h16* Wc[4][4];
    int kdp[4][4];
    size_t woff = 0;
    for (int blk = 0; blk < 4; ++blk) {
        kdp[blk][0] = (blk == 0) ? 224 : 448;
        kdp[blk][1] = 320;
        kdp[blk][2] = 192;
        kdp[blk][3] = (blk == 0) ? 32 : 0;
        for (int j = 0; j < 4; ++j) {
            Wc[blk][j] = wbase + woff;
            woff += (size_t)64 * kdp[blk][j];
        }
    }

    // ---- zero BN stat accumulators (atomics add onto them) ----
    hipMemsetAsync(stats, 0, 16 * 4 * 2048 * sizeof(float), stream);

    // ---- weight conversion ----
    auto cvtW = [&](const float* w, h16* A, int cin, int cinPad, int K, int kp) {
        int n = 64 * kp;
        cvt_weight<<<(n + 255) / 256, 256, 0, stream>>>(w, A, cin, cinPad, K, kp);
    };
    cvtW(bp[0].wx, Wc[0][0], CINV, 32, 7, 224);
    cvtW(bp[0].wy, Wc[0][1], 64,   64, 5, 320);
    cvtW(bp[0].wz, Wc[0][2], 64,   64, 3, 192);
    cvtW(bp[0].ws, Wc[0][3], CINV, 32, 1, 32);
    for (int blk = 1; blk < 4; ++blk) {
        cvtW(bp[blk].wx, Wc[blk][0], 64, 64, 7, 448);
        cvtW(bp[blk].wy, Wc[blk][1], 64, 64, 5, 320);
        cvtW(bp[blk].wz, Wc[blk][2], 64, 64, 3, 192);
    }

    // ---- data fp32 -> fp16 ----
    {
        int n = BB * NN * CINV * LL;
        cvt_data<<<(n + 255) / 256, 256, 0, stream>>>(data, P[0], n);
    }

    const int RES_BLOCKS = (int)(((size_t)MT * 64 * LL / 8 + 255) / 256);

    // ---- block 1 (expand: conv shortcut) ----
    {
        int s0 = 0;
        conv_wmma<7,5,7><<<MT, 256, 0, stream>>>(P[0], CINV, nullptr, nullptr, 0, 0,
                                          Wc[0][0], bp[0].bx, P[1], SUM(s0), SQ(s0));
        bn_finalize<<<8, 256, 0, stream>>>(SUM(s0), SQ(s0), bp[0].gx, bp[0].bex, SC(s0), SH(s0));
        conv_wmma<5,6,10><<<MT, 256, 0, stream>>>(P[1], 64, SC(s0), SH(s0), 1, 1,
                                          Wc[0][1], bp[0].by, P[2], SUM(s0+1), SQ(s0+1));
        bn_finalize<<<8, 256, 0, stream>>>(SUM(s0+1), SQ(s0+1), bp[0].gy, bp[0].bey, SC(s0+1), SH(s0+1));
        conv_wmma<3,6,6><<<MT, 256, 0, stream>>>(P[2], 64, SC(s0+1), SH(s0+1), 1, 1,
                                          Wc[0][2], bp[0].bz, P[1], SUM(s0+2), SQ(s0+2));
        bn_finalize<<<8, 256, 0, stream>>>(SUM(s0+2), SQ(s0+2), bp[0].gz, bp[0].bez, SC(s0+2), SH(s0+2));
        // shortcut: 1x1 conv on the raw input
        conv_wmma<1,5,1><<<MT, 256, 0, stream>>>(P[0], CINV, nullptr, nullptr, 0, 0,
                                          Wc[0][3], bp[0].bs, P[3], SUM(s0+3), SQ(s0+3));
        bn_finalize<<<8, 256, 0, stream>>>(SUM(s0+3), SQ(s0+3), bp[0].gs, bp[0].bes, SC(s0+3), SH(s0+3));
        residual_add<<<RES_BLOCKS, 256, 0, stream>>>(P[1], P[3], SC(s0+2), SH(s0+2),
                                                     SC(s0+3), SH(s0+3), P[2]);
    }

    // ---- blocks 2..4 (identity shortcut) ----
    int x = 2;   // index of current activation buffer
    for (int blk = 1; blk < 4; ++blk) {
        int s0 = blk * 4;
        int a = 0;
        int bb = (x == 1) ? 2 : 1;
        conv_wmma<7,6,14><<<MT, 256, 0, stream>>>(P[x], 64, nullptr, nullptr, 0, 0,
                                          Wc[blk][0], bp[blk].bx, P[a], SUM(s0), SQ(s0));
        bn_finalize<<<8, 256, 0, stream>>>(SUM(s0), SQ(s0), bp[blk].gx, bp[blk].bex, SC(s0), SH(s0));
        conv_wmma<5,6,10><<<MT, 256, 0, stream>>>(P[a], 64, SC(s0), SH(s0), 1, 1,
                                          Wc[blk][1], bp[blk].by, P[bb], SUM(s0+1), SQ(s0+1));
        bn_finalize<<<8, 256, 0, stream>>>(SUM(s0+1), SQ(s0+1), bp[blk].gy, bp[blk].bey, SC(s0+1), SH(s0+1));
        conv_wmma<3,6,6><<<MT, 256, 0, stream>>>(P[bb], 64, SC(s0+1), SH(s0+1), 1, 1,
                                          Wc[blk][2], bp[blk].bz, P[a], SUM(s0+2), SQ(s0+2));
        bn_finalize<<<8, 256, 0, stream>>>(SUM(s0+2), SQ(s0+2), bp[blk].gz, bp[blk].bez, SC(s0+2), SH(s0+2));
        // identity-shortcut BN stats straight from the block input
        act_stats<<<BB * 64, 256, 0, stream>>>(P[x], SUM(s0+3), SQ(s0+3));
        bn_finalize<<<8, 256, 0, stream>>>(SUM(s0+3), SQ(s0+3), bp[blk].gs, bp[blk].bes, SC(s0+3), SH(s0+3));
        residual_add<<<RES_BLOCKS, 256, 0, stream>>>(P[a], P[x], SC(s0+2), SH(s0+2),
                                                     SC(s0+3), SH(s0+3), P[bb]);
        x = bb;
    }

    // ---- pool + GCN head ----
    avg_pool<<<(MT * 64) / 256, 256, 0, stream>>>(P[x], pooled);
    gcn_head<<<BB, 64, 0, stream>>>(pooled, adj, gcw, gcb, (float*)d_out);
}
